// TrulyNeuralRegisterFile_18975165514062
// MI455X (gfx1250) — matrix operand order repaired and checked by reading, not compiled, not run on hardware
//
#include <hip/hip_runtime.h>

// ---------------------------------------------------------------------------
// TrulyNeuralRegisterFile — MI455X (gfx1250, wave32, WMMA) fused implementation
//
// Key insight: attend(idx) has only 32 distinct outputs (idx in [0,32)), so the
// query-encoder MLP + softmax collapse to 32-entry tables, and att_w.mean(0)
// collapses to a histogram. Only the enc/erase/add MLPs are true per-batch
// GEMMs (~12.9 GFLOP) -> run them as f16 WMMA with fp32 accumulation.
// Input tiles are staged with global_load_async_to_lds_b128 (ASYNCcnt path).
// ---------------------------------------------------------------------------

typedef __attribute__((ext_vector_type(16))) _Float16 v16h;
typedef __attribute__((ext_vector_type(8)))  _Float16 v8h;
typedef __attribute__((ext_vector_type(8)))  float    v8f;

__device__ __forceinline__ float gelu_exact(float x) {
    return 0.5f * x * (1.0f + erff(x * 0.70710678118654752f));
}

// Generic pointers into LDS carry the byte offset in addr[31:0] (LDS aperture
// truncates), so this yields the VDST LDS address for async-to-LDS ops.
__device__ __forceinline__ unsigned lds_off(const void* p) {
    return (unsigned)(unsigned long long)p;
}

// Issue one 16-byte async global->LDS copy (tracked by ASYNCcnt).
__device__ __forceinline__ void async_copy_b128(const void* gptr,
                                                const void* lptr) {
    asm volatile("global_load_async_to_lds_b128 %0, %1, off"
                 :: "v"(lds_off(lptr)), "v"(gptr)
                 : "memory");
}

__device__ __forceinline__ void wait_async_all() {
    asm volatile("s_wait_asynccnt 0x0" ::: "memory");
}

// ---------------------------------------------------------------------------
// WMMA operand helpers (wave32, 16x16x32 f16).
// A (16x32 f16): lane<16 holds row M=lane, K halves {k0..k0+7, k0+16..k0+23};
//                lane>=16 holds row M=lane-16, K halves {k0+8..15, k0+24..31}.
// B is pre-packed per (nt,kt) tile: 32 lanes x 16 contiguous halves.
// C/D (16x16 f32): VGPR i, lane l -> row = 8*(l>>4)+i, col = l&15.
// ---------------------------------------------------------------------------
__device__ __forceinline__ v16h ldA(const _Float16* s, int ld, int mbase,
                                    int kbase, int lane) {
    const int hi = lane >> 4;
    const _Float16* p = s + (mbase + (lane & 15)) * ld + kbase + 8 * hi;
    v8h a0 = *(const v8h*)(p);
    v8h a1 = *(const v8h*)(p + 16);
    v16h r;
#pragma unroll
    for (int i = 0; i < 8; i++) { r[i] = a0[i]; r[i + 8] = a1[i]; }
    return r;
}

__device__ __forceinline__ v16h ldB(const _Float16* gW, int KT, int nt, int kt,
                                    int lane) {
    const _Float16* p = gW + (((nt * KT) + kt) * 32 + lane) * 16;
    v8h b0 = *(const v8h*)(p);
    v8h b1 = *(const v8h*)(p + 8);
    v16h r;
#pragma unroll
    for (int i = 0; i < 8; i++) { r[i] = b0[i]; r[i + 8] = b1[i]; }
    return r;
}

__device__ __forceinline__ void stC(float* s, int ld, int mbase, int nbase,
                                    int lane, v8f c) {
    float* p = s + (mbase + 8 * (lane >> 4)) * ld + nbase + (lane & 15);
#pragma unroll
    for (int i = 0; i < 8; i++) p[i * ld] = c[i];
}

// One wave computes a 16-row strip x NTW 16-col tiles, K = KT*32.
template <int KT, int NTW>
__device__ __forceinline__ void wave_gemm(const _Float16* sAct, int lda,
                                          const _Float16* gW, float* sOut,
                                          int ldo, int mbase, int ntbase,
                                          int lane) {
    v8f acc[NTW];
#pragma unroll
    for (int n = 0; n < NTW; n++) {
#pragma unroll
        for (int i = 0; i < 8; i++) acc[n][i] = 0.0f;
    }
#pragma unroll
    for (int kt = 0; kt < KT; kt++) {
        v16h a = ldA(sAct, lda, mbase, kt * 32, lane);
#pragma unroll
        for (int n = 0; n < NTW; n++) {
            v16h b = ldB(gW, KT, ntbase + n, kt, lane);
            acc[n] = __builtin_amdgcn_wmma_f32_16x16x32_f16(
                false, a, false, b, (short)0, acc[n], false, false);
        }
    }
#pragma unroll
    for (int n = 0; n < NTW; n++)
        stC(sOut, ldo, mbase, (ntbase + n) * 16, lane, acc[n]);
}

// ---------------------------------------------------------------------------
// Kernel: zero accumulators (hist + e_sum + a_sum) each launch.
// ---------------------------------------------------------------------------
__global__ void zero_kernel(int* hist, float* e_sum, float* a_sum) {
    const int t = threadIdx.x;
    if (t < 32) hist[t] = 0;
    if (t < 64) { e_sum[t] = 0.0f; a_sum[t] = 0.0f; }
}

// ---------------------------------------------------------------------------
// Kernel: histogram of write_idx (32 bins).
// ---------------------------------------------------------------------------
__global__ __launch_bounds__(256) void hist_kernel(const int* __restrict__ widx,
                                                   int* __restrict__ hist,
                                                   int B) {
    __shared__ int lh[32];
    if (threadIdx.x < 32) lh[threadIdx.x] = 0;
    __syncthreads();
    const int stride = gridDim.x * blockDim.x;
    for (int i = blockIdx.x * blockDim.x + threadIdx.x; i < B; i += stride)
        atomicAdd(&lh[widx[i] & 31], 1);
    __syncthreads();
    if (threadIdx.x < 32) atomicAdd(&hist[threadIdx.x], lh[threadIdx.x]);
}

// ---------------------------------------------------------------------------
// Kernel: precompute the 32-query tables (att_tab, qg_eg, qg_wg). One block.
// ---------------------------------------------------------------------------
__device__ __forceinline__ float block_sum(float v, float* red) {
    const int t = threadIdx.x;
    red[t] = v;
    __syncthreads();
    for (int s = 128; s > 0; s >>= 1) {
        if (t < s) red[t] += red[t + s];
        __syncthreads();
    }
    float r = red[0];
    __syncthreads();
    return r;
}

__global__ __launch_bounds__(256) void precompute_kernel(
    const float* __restrict__ rkeys, const float* __restrict__ temp,
    const float* __restrict__ qe_w1, const float* __restrict__ qe_b1,
    const float* __restrict__ qe_g1, const float* __restrict__ qe_bt1,
    const float* __restrict__ qe_w2, const float* __restrict__ qe_b2,
    const float* __restrict__ qe_g2, const float* __restrict__ qe_bt2,
    const float* __restrict__ qe_w3, const float* __restrict__ qe_b3,
    const float* __restrict__ eg_w1, const float* __restrict__ eg_b1,
    const float* __restrict__ wg_w1, const float* __restrict__ wg_b1,
    float* __restrict__ att, float* __restrict__ qge, float* __restrict__ qgw) {
    __shared__ float kQ[32 * 256];   // raw queries for all 32 indices
    __shared__ float kS[32 * 32];
    __shared__ float kRed[256];
    __shared__ float kInvK[32];
    __shared__ float kInvQ[32];
    __shared__ float kH1[128];
    __shared__ float kH2[256];
    const int t = threadIdx.x;

    // inverse norms of register keys
    if (t < 32) {
        float s = 0.0f;
        for (int k = 0; k < 256; k++) { float v = rkeys[t * 256 + k]; s += v * v; }
        kInvK[t] = 1.0f / fmaxf(sqrtf(s), 1e-12f);
    }
    __syncthreads();

    // query MLP for each of the 32 possible indices
    for (int i = 0; i < 32; i++) {
        float pre = 0.0f;
        if (t < 128) {
            pre = qe_b1[t];
#pragma unroll
            for (int k = 0; k < 5; k++)
                if ((i >> k) & 1) pre += qe_w1[k * 128 + t];
        }
        float s1 = block_sum(t < 128 ? pre : 0.0f, kRed);
        float s2 = block_sum(t < 128 ? pre * pre : 0.0f, kRed);
        if (t < 128) {
            float mean = s1 * (1.0f / 128.0f);
            float var = s2 * (1.0f / 128.0f) - mean * mean;
            float y = (pre - mean) * rsqrtf(var + 1e-5f) * qe_g1[t] + qe_bt1[t];
            kH1[t] = gelu_exact(y);
        }
        __syncthreads();

        float pre2 = qe_b2[t];
        for (int k = 0; k < 128; k++) pre2 += kH1[k] * qe_w2[k * 256 + t];
        float u1 = block_sum(pre2, kRed);
        float u2 = block_sum(pre2 * pre2, kRed);
        {
            float mean = u1 * (1.0f / 256.0f);
            float var = u2 * (1.0f / 256.0f) - mean * mean;
            float y = (pre2 - mean) * rsqrtf(var + 1e-5f) * qe_g2[t] + qe_bt2[t];
            kH2[t] = gelu_exact(y);
        }
        __syncthreads();

        float q = qe_b3[t];
        for (int k = 0; k < 256; k++) q += kH2[k] * qe_w3[k * 256 + t];
        kQ[i * 256 + t] = q;
        __syncthreads();
    }

    if (t < 32) {
        float s = 0.0f;
        for (int k = 0; k < 256; k++) { float v = kQ[t * 256 + k]; s += v * v; }
        kInvQ[t] = 1.0f / fmaxf(sqrtf(s), 1e-12f);
    }
    __syncthreads();

    // attention table (32x32) with softmax rows
    const float invT = 1.0f / fmaxf(fabsf(temp[0]), 0.01f);
    for (int idx = t; idx < 1024; idx += 256) {
        int i = idx >> 5, j = idx & 31;
        float s = 0.0f;
        for (int k = 0; k < 256; k++) s += kQ[i * 256 + k] * rkeys[j * 256 + k];
        kS[idx] = s * kInvQ[i] * kInvK[j] * invT;
    }
    __syncthreads();
    if (t < 32) {
        float m = -1e30f;
        for (int j = 0; j < 32; j++) m = fmaxf(m, kS[t * 32 + j]);
        float sum = 0.0f;
        for (int j = 0; j < 32; j++) sum += expf(kS[t * 32 + j] - m);
        float inv = 1.0f / sum;
        for (int j = 0; j < 32; j++) att[t * 32 + j] = expf(kS[t * 32 + j] - m) * inv;
    }

    // q contributions to gate layer 1 (+ biases folded in)
    for (int idx = t; idx < 4096; idx += 256) {
        int i = idx >> 7, c = idx & 127;
        float se = eg_b1[c], sw = wg_b1[c];
        for (int k = 0; k < 256; k++) {
            float qv = kQ[i * 256 + k];
            se += qv * eg_w1[(64 + k) * 128 + c];
            sw += qv * wg_w1[(64 + k) * 128 + c];
        }
        qge[idx] = se;
        qgw[idx] = sw;
    }
}

// ---------------------------------------------------------------------------
// Kernel: pack the six GEMM weight matrices into f16 WMMA B-operand layout.
// packed[((nt*KT + kt)*32 + lane)*16 + j] =
//     W[kt*32 + 8*(lane>>4) + (j&7) + 16*(j>>3)][nt*16 + (lane&15)]
// Each matrix packs to exactly 8192 halves (16KB).
// ---------------------------------------------------------------------------
__global__ __launch_bounds__(256) void pack_weights_kernel(
    const float* __restrict__ ve_w1, const float* __restrict__ ve_w2,
    const float* __restrict__ eg_w1, const float* __restrict__ eg_w2,
    const float* __restrict__ wg_w1, const float* __restrict__ wg_w2,
    _Float16* __restrict__ pk) {
    const int gid = blockIdx.x * 256 + threadIdx.x;  // 0..49151
    const int m = gid >> 13;
    const int p = gid & 8191;
    const float* W;
    int K, N;
    switch (m) {
        case 0: W = ve_w1; K = 64;  N = 128; break;
        case 1: W = ve_w2; K = 128; N = 64;  break;
        case 2: W = eg_w1; K = 64;  N = 128; break;  // top 64 rows of [320,128]
        case 3: W = eg_w2; K = 128; N = 64;  break;
        case 4: W = wg_w1; K = 64;  N = 128; break;  // top 64 rows of [320,128]
        default: W = wg_w2; K = 128; N = 64; break;
    }
    const int KT = K >> 5;
    const int j = p & 15;
    const int lane = (p >> 4) & 31;
    const int rest = p >> 9;
    const int kt = rest % KT;
    const int nt = rest / KT;
    const int k = kt * 32 + ((lane >> 4) << 3) + (j & 7) + ((j >> 3) << 4);
    const int n = nt * 16 + (lane & 15);
    pk[m * 8192 + p] = (_Float16)W[k * N + n];
}

// ---------------------------------------------------------------------------
// Main fused kernel: 64 batch rows per block, 8 waves.
// wave -> (mtile = wid&3, nhalf = wid>>2): 4 M-strips x 2 N-halves.
// Chain: X -> GEMM(ve1) -> LN+GELU -> GEMM(ve2)+b2 = enc ->
//        [GEMM(eg1)+qge -> GELU -> GEMM(eg2) -> sigmoid -> col-sum atomics]
//        [GEMM(wg1)+qgw -> GELU -> GEMM(wg2) ->            col-sum atomics]
// 48 v_wmma_f32_16x16x32_f16 per wave per block.
// ---------------------------------------------------------------------------
__global__ __launch_bounds__(256, 1) void main_fused_kernel(
    const float* __restrict__ wval, const int* __restrict__ widx,
    const _Float16* __restrict__ pk, const float* __restrict__ qge,
    const float* __restrict__ qgw, const float* __restrict__ ve_b1,
    const float* __restrict__ ve_g1, const float* __restrict__ ve_bt1,
    const float* __restrict__ ve_b2, const float* __restrict__ eg_b2,
    const float* __restrict__ wg_b2, float* __restrict__ e_sum,
    float* __restrict__ a_sum, int B) {
    __shared__ __align__(16) float    sH[64 * 128];   // f32 GEMM outputs
    __shared__ __align__(16) _Float16 sA[64 * 128];   // f16 128-wide activations
    __shared__ __align__(16) _Float16 sX[64 * 64];    // f16 64-wide (X, then enc)
    __shared__ int   sWI[64];
    __shared__ float sB1[128], sG1[128], sBt1[128];
    __shared__ float sB2[64], sEB2[64], sWB2[64];

    const int t = threadIdx.x;
    const int lane = t & 31;
    const int wid = t >> 5;
    const int mbase = (wid & 3) * 16;
    const int nhalf = wid >> 2;
    const size_t row0 = (size_t)blockIdx.x * 64;

    // ---- stage raw f32 input tile (16KB) into sH via async-to-LDS DMA ----
    // 1024 x 16B chunks, 4 per thread; tracked by ASYNCcnt, overlapped with
    // the parameter loads below.
#pragma unroll
    for (int k = 0; k < 4; k++) {
        const int chunk = k * 256 + t;
        async_copy_b128(wval + row0 * 64 + (size_t)chunk * 4, &sH[chunk * 4]);
    }
    if (t < 64) sWI[t] = widx[row0 + t] & 31;
    if (t < 128) { sB1[t] = ve_b1[t]; sG1[t] = ve_g1[t]; sBt1[t] = ve_bt1[t]; }
    if (t < 64)  { sB2[t] = ve_b2[t]; sEB2[t] = eg_b2[t]; sWB2[t] = wg_b2[t]; }
    if ((int)row0 + 64 < B)  // prefetch next tile (global_prefetch_b8)
        __builtin_prefetch(wval + (row0 + 64) * 64 + t * 16, 0, 0);
    wait_async_all();
    __syncthreads();
    // f32 -> f16 conversion into the A-operand staging buffer
    for (int i = t; i < 4096; i += 256) sX[i] = (_Float16)sH[i];
    __syncthreads();

    // ---- GEMM1: X(64x64) @ ve_w1(64x128) -> sH ----
    wave_gemm<2, 4>(sX, 64, pk + 0 * 8192, sH, 128, mbase, nhalf * 4, lane);
    __syncthreads();

    // ---- LayerNorm + GELU -> sA (f16) ; 4 lanes per row ----
    {
        const int r = t >> 2, qq = t & 3, c0 = qq * 32;
        float sum = 0.0f, ss = 0.0f;
        for (int c = c0; c < c0 + 32; c++) {
            float x = sH[r * 128 + c] + sB1[c];
            sum += x; ss += x * x;
        }
        sum += __shfl_xor(sum, 1); ss += __shfl_xor(ss, 1);
        sum += __shfl_xor(sum, 2); ss += __shfl_xor(ss, 2);
        const float mean = sum * (1.0f / 128.0f);
        const float var = ss * (1.0f / 128.0f) - mean * mean;
        const float inv = rsqrtf(var + 1e-5f);
        for (int c = c0; c < c0 + 32; c++) {
            float x = sH[r * 128 + c] + sB1[c];
            sA[r * 128 + c] =
                (_Float16)gelu_exact((x - mean) * inv * sG1[c] + sBt1[c]);
        }
    }
    __syncthreads();

    // ---- GEMM2: sA(64x128) @ ve_w2(128x64) -> sH; enc = +b2 -> sX ----
    wave_gemm<4, 2>(sA, 128, pk + 1 * 8192, sH, 64, mbase, nhalf * 2, lane);
    __syncthreads();
    for (int i = t; i < 4096; i += 256)
        sX[i] = (_Float16)(sH[i] + sB2[i & 63]);
    __syncthreads();

    // ---- erase branch ----
    wave_gemm<2, 4>(sX, 64, pk + 2 * 8192, sH, 128, mbase, nhalf * 4, lane);
    __syncthreads();
    for (int i = t; i < 8192; i += 256) {
        int r = i >> 7, c = i & 127;
        sA[i] = (_Float16)gelu_exact(sH[i] + qge[sWI[r] * 128 + c]);
    }
    __syncthreads();
    wave_gemm<4, 2>(sA, 128, pk + 3 * 8192, sH, 64, mbase, nhalf * 2, lane);
    __syncthreads();
    {
        const int c = t & 63, qq = t >> 6;
        float s = 0.0f;
        for (int r = qq * 16; r < qq * 16 + 16; r++) {
            float x = sH[r * 64 + c] + sEB2[c];
            s += 1.0f / (1.0f + expf(-x));
        }
        atomicAdd(&e_sum[c], s);
    }
    __syncthreads();

    // ---- add branch ----
    wave_gemm<2, 4>(sX, 64, pk + 4 * 8192, sH, 128, mbase, nhalf * 4, lane);
    __syncthreads();
    for (int i = t; i < 8192; i += 256) {
        int r = i >> 7, c = i & 127;
        sA[i] = (_Float16)gelu_exact(sH[i] + qgw[sWI[r] * 128 + c]);
    }
    __syncthreads();
    wave_gemm<4, 2>(sA, 128, pk + 5 * 8192, sH, 64, mbase, nhalf * 2, lane);
    __syncthreads();
    {
        const int c = t & 63, qq = t >> 6;
        float s = 0.0f;
        for (int r = qq * 16; r < qq * 16 + 16; r++)
            s += sH[r * 64 + c] + sWB2[c];
        atomicAdd(&a_sum[c], s);
    }
}

// ---------------------------------------------------------------------------
// Kernel: finalize memory update + build read table. One block.
// ---------------------------------------------------------------------------
__global__ __launch_bounds__(256) void finalize_kernel(
    const float* __restrict__ memory, const float* __restrict__ att,
    const int* __restrict__ hist, const float* __restrict__ e_sum,
    const float* __restrict__ a_sum, float invB,
    float* __restrict__ out_tail, float* __restrict__ rtab) {
    __shared__ float satt[1024], snm[2048];
    __shared__ float wbar[32], ebar[64], abar[64];
    const int t = threadIdx.x;
    for (int i = t; i < 1024; i += 256) satt[i] = att[i];
    __syncthreads();
    if (t < 32) {
        float s = 0.0f;
        for (int j = 0; j < 32; j++)
            if (j != 31) s += (float)hist[j] * satt[j * 32 + t];
        wbar[t] = s * invB;
    }
    if (t < 64) { ebar[t] = e_sum[t] * invB; abar[t] = a_sum[t] * invB; }
    __syncthreads();
    for (int i = t; i < 2048; i += 256) {
        int r = i >> 6, c = i & 63;
        float nm = memory[i] * (1.0f - wbar[r] * ebar[c]) + wbar[r] * abar[c];
        snm[i] = nm;
        out_tail[i] = nm;  // new_memory output
    }
    __syncthreads();
    for (int i = t; i < 2048; i += 256) {
        int j = i >> 6, c = i & 63;
        float s = 0.0f;
        for (int k = 0; k < 32; k++) s += satt[j * 32 + k] * snm[k * 64 + c];
        rtab[i] = (j == 31) ? 0.0f : s;
    }
}

// ---------------------------------------------------------------------------
// Kernel: read_value[b] = read_tab[read_idx[b]]  (pure bandwidth, float4).
// read table broadcast into LDS via async-to-LDS DMA.
// ---------------------------------------------------------------------------
__global__ __launch_bounds__(256) void scatter_read_kernel(
    const int* __restrict__ ridx, const float* __restrict__ rtab,
    float* __restrict__ out) {
    __shared__ __align__(16) float s[2048];
#pragma unroll
    for (int k = 0; k < 2; k++) {
        const int chunk = k * 256 + threadIdx.x;  // 512 x 16B = 8KB
        async_copy_b128(rtab + chunk * 4, &s[chunk * 4]);
    }
    wait_async_all();
    __syncthreads();
    const int base = blockIdx.x * 1024;
#pragma unroll
    for (int k = 0; k < 4; k++) {
        int idx = base + k * 256 + threadIdx.x;  // float4 index
        int b = idx >> 4, seg = idx & 15;
        int r = ridx[b] & 31;
        float4 v = *(const float4*)(&s[r * 64 + seg * 4]);
        ((float4*)out)[idx] = v;
    }
}

// ---------------------------------------------------------------------------
// Host launcher
// ---------------------------------------------------------------------------
extern "C" void kernel_launch(void* const* d_in, const int* in_sizes, int n_in,
                              void* d_out, int out_size, void* d_ws,
                              size_t ws_size, hipStream_t stream) {
    const float* memory = (const float*)d_in[0];
    const int*   widx   = (const int*)d_in[1];
    const float* wval   = (const float*)d_in[2];
    const int*   ridx   = (const int*)d_in[3];
    const float* rkeys  = (const float*)d_in[4];
    const float* temp   = (const float*)d_in[5];
    const float* qe_w1  = (const float*)d_in[6];
    const float* qe_b1  = (const float*)d_in[7];
    const float* qe_g1  = (const float*)d_in[8];
    const float* qe_bt1 = (const float*)d_in[9];
    const float* qe_w2  = (const float*)d_in[10];
    const float* qe_b2  = (const float*)d_in[11];
    const float* qe_g2  = (const float*)d_in[12];
    const float* qe_bt2 = (const float*)d_in[13];
    const float* qe_w3  = (const float*)d_in[14];
    const float* qe_b3  = (const float*)d_in[15];
    const float* ve_w1  = (const float*)d_in[16];
    const float* ve_b1  = (const float*)d_in[17];
    const float* ve_g1  = (const float*)d_in[18];
    const float* ve_bt1 = (const float*)d_in[19];
    const float* ve_w2  = (const float*)d_in[20];
    const float* ve_b2  = (const float*)d_in[21];
    const float* eg_w1  = (const float*)d_in[22];
    const float* eg_b1  = (const float*)d_in[23];
    const float* eg_w2  = (const float*)d_in[24];
    const float* eg_b2  = (const float*)d_in[25];
    const float* wg_w1  = (const float*)d_in[26];
    const float* wg_b1  = (const float*)d_in[27];
    const float* wg_w2  = (const float*)d_in[28];
    const float* wg_b2  = (const float*)d_in[29];
    float* out = (float*)d_out;
    const int B = in_sizes[1];  // 131072

    // workspace carve (bytes)
    char* ws = (char*)d_ws;
    float*    att  = (float*)(ws + 0);        //  4096 B : att_tab[32][32]
    float*    qge  = (float*)(ws + 4096);     // 16384 B : qg_eg[32][128]
    float*    qgw  = (float*)(ws + 20480);    // 16384 B : qg_wg[32][128]
    _Float16* pk   = (_Float16*)(ws + 36864); // 98304 B : 6 packed f16 matrices
    int*      hist = (int*)(ws + 135168);     //   128 B
    float*    esum = (float*)(ws + 135296);   //   256 B
    float*    asum = (float*)(ws + 135552);   //   256 B
    float*    rtab = (float*)(ws + 135808);   //  8192 B : read_tab[32][64]

    zero_kernel<<<1, 64, 0, stream>>>(hist, esum, asum);
    hist_kernel<<<64, 256, 0, stream>>>(widx, hist, B);
    precompute_kernel<<<1, 256, 0, stream>>>(
        rkeys, temp, qe_w1, qe_b1, qe_g1, qe_bt1, qe_w2, qe_b2, qe_g2, qe_bt2,
        qe_w3, qe_b3, eg_w1, eg_b1, wg_w1, wg_b1, att, qge, qgw);
    pack_weights_kernel<<<192, 256, 0, stream>>>(ve_w1, ve_w2, eg_w1, eg_w2,
                                                 wg_w1, wg_w2, pk);
    main_fused_kernel<<<B / 64, 256, 0, stream>>>(
        wval, widx, pk, qge, qgw, ve_b1, ve_g1, ve_bt1, ve_b2, eg_b2, wg_b2,
        esum, asum, B);
    finalize_kernel<<<1, 256, 0, stream>>>(memory, att, hist, esum, asum,
                                           1.0f / (float)B,
                                           out + (size_t)B * 64, rtab);
    scatter_read_kernel<<<(B * 16) / 1024, 256, 0, stream>>>(ridx, rtab, out);
}
